// Attention_86062554677427
// MI455X (gfx1250) — matrix operand order repaired and checked by reading, not compiled, not run on hardware
//
#include <hip/hip_runtime.h>

typedef __attribute__((ext_vector_type(16))) _Float16 v16h;
typedef __attribute__((ext_vector_type(8)))  _Float16 v8h;
typedef __attribute__((ext_vector_type(4)))  _Float16 v4h;
typedef __attribute__((ext_vector_type(8)))  float    v8f;

#define B_     2
#define N_     2048
#define D_     1024
#define H_     16
#define HD_    64
#define INNER_ 1024

#define LDA 40      // halfs; 80B row stride -> 16B aligned, padded
#define LDB 40
#define MAXT 18
#define NEG_LIM (-65504.0f)

__device__ __forceinline__ v8f wmma_f32_f16(v16h a, v16h b, v8f c) {
  return __builtin_amdgcn_wmma_f32_16x16x32_f16(false, a, false, b, (short)0, c,
                                                false, false);
}

__device__ __forceinline__ v16h cat8(v8h lo, v8h hi) {
  return __builtin_shufflevector(lo, hi, 0, 1, 2, 3, 4, 5, 6, 7,
                                 8, 9, 10, 11, 12, 13, 14, 15);
}

// ---------------------------------------------------------------------------
// Kernel 1: fused QKV projection + bias + RoPE (first 64 inner channels only,
// matching the reference's rot = freqs.shape[-1] = 64 behavior).
// ---------------------------------------------------------------------------
__global__ __launch_bounds__(256)
void qkv_rope_kernel(const float* __restrict__ x,
                     const float* __restrict__ Wq, const float* __restrict__ bq,
                     const float* __restrict__ Wk, const float* __restrict__ bk,
                     const float* __restrict__ Wv, const float* __restrict__ bv,
                     const float* __restrict__ freqs,
                     _Float16* __restrict__ qh, _Float16* __restrict__ kh,
                     _Float16* __restrict__ vt)
{
  __shared__ __align__(16) _Float16 As[64 * LDA];
  __shared__ __align__(16) _Float16 Bs[128 * LDB];

  const int mat = blockIdx.z;
  const float* __restrict__ W    = (mat == 0) ? Wq : (mat == 1) ? Wk : Wv;
  const float* __restrict__ bias = (mat == 0) ? bq : (mat == 1) ? bk : bv;

  const int row0 = blockIdx.x * 64;
  const int col0 = blockIdx.y * 128;
  const int tid  = threadIdx.x;
  const int wave = tid >> 5;
  const int lane = tid & 31;
  const int hg   = lane >> 4;
  const int nn   = lane & 15;
  const int wm   = wave >> 1;
  const int wn   = wave & 1;

  // B staging micro-tile: 4 k-rows x 4 n-cols per thread
  const int bk_ = (tid >> 5) * 4;          // k:  0..28
  const int bn_ = (tid & 31) * 4;          // n:  0..124

  v8f acc[4] = {};

  for (int k0 = 0; k0 < D_; k0 += 32) {
    __syncthreads();
    // stage A: x tile 64x32, f32 -> f16, [m][k], vector store
    {
      const int e = tid * 8;
      const int r = e >> 5, c = e & 31;
      const float* src = x + (size_t)(row0 + r) * D_ + k0 + c;
      float4 v0 = *(const float4*)(src);
      float4 v1 = *(const float4*)(src + 4);
      v8h pk = {(_Float16)v0.x, (_Float16)v0.y, (_Float16)v0.z, (_Float16)v0.w,
                (_Float16)v1.x, (_Float16)v1.y, (_Float16)v1.z, (_Float16)v1.w};
      *(v8h*)&As[r * LDA + c] = pk;
    }
    // stage B: W tile 32x128 f32 -> f16 transposed [n][k]; 4xb64 LDS stores
    {
      float rr[4][4];
      #pragma unroll
      for (int i = 0; i < 4; ++i) {
        float4 v = *(const float4*)(W + (size_t)(k0 + bk_ + i) * INNER_ + col0 + bn_);
        rr[i][0] = v.x; rr[i][1] = v.y; rr[i][2] = v.z; rr[i][3] = v.w;
      }
      #pragma unroll
      for (int j = 0; j < 4; ++j) {
        v4h pk = {(_Float16)rr[0][j], (_Float16)rr[1][j],
                  (_Float16)rr[2][j], (_Float16)rr[3][j]};
        *(v4h*)&Bs[(bn_ + j) * LDB + bk_] = pk;
      }
    }
    if (k0 + 32 < D_) {
      __builtin_prefetch(W + (size_t)(k0 + 32) * INNER_ + col0, 0, 3);
      __builtin_prefetch(x + (size_t)row0 * D_ + k0 + 32, 0, 3);
    }
    __syncthreads();

    // fragments
    const _Float16* pa = &As[(wm * 16 + nn) * LDA + hg * 8];
    const v16h afr = cat8(*(const v8h*)pa, *(const v8h*)(pa + 16));
    #pragma unroll
    for (int a4 = 0; a4 < 4; ++a4) {
      const _Float16* pb = &Bs[(wn * 64 + a4 * 16 + nn) * LDB + hg * 8];
      const v16h bfr = cat8(*(const v8h*)pb, *(const v8h*)(pb + 16));
      acc[a4] = wmma_f32_f16(afr, bfr, acc[a4]);
    }
  }

  #pragma unroll
  for (int a4 = 0; a4 < 4; ++a4) {
    const int  C  = col0 + wn * 64 + a4 * 16 + nn;
    const float bc = bias[C];
    const bool do_rope = (mat < 2) && (C < HD_);   // wave-uniform
    #pragma unroll
    for (int r = 0; r < 8; ++r) {
      const int R   = row0 + r + 8 * hg;
      const int bI  = R >> 11;
      const int tok = R & (N_ - 1);
      float val = acc[a4][r] + bc;
      if (do_rope) {
        const float partner = __shfl_xor(val, 1, 32);
        const float f  = freqs[(size_t)tok * HD_ + C];
        const float cf = __cosf(f), sf = __sinf(f);
        val = (C & 1) ? (val * cf + partner * sf)
                      : (val * cf - partner * sf);
      }
      const int hIdx = C >> 6, dIdx = C & 63;
      const size_t bh = (size_t)bI * H_ + hIdx;
      if (mat == 0)      qh[(bh * N_ + tok) * HD_ + dIdx] = (_Float16)val;
      else if (mat == 1) kh[(bh * N_ + tok) * HD_ + dIdx] = (_Float16)val;
      else               vt[(bh * HD_ + dIdx) * N_ + tok] = (_Float16)val;
    }
  }
}

// ---------------------------------------------------------------------------
// Kernel 2: windowed attention, one wave per 16-query block of one (b,h).
// ---------------------------------------------------------------------------
__global__ __launch_bounds__(128)
void attn_kernel(const _Float16* __restrict__ qh, const _Float16* __restrict__ kh,
                 const _Float16* __restrict__ vt,
                 const unsigned char* __restrict__ mask,
                 const int* __restrict__ wsz,
                 _Float16* __restrict__ ao)
{
  __shared__ __align__(16) _Float16 sS[4][16][MAXT * 16];
  __shared__ float sMax[4][16];
  __shared__ float sSum[4][16];

  const int tid  = threadIdx.x;
  const int wave = tid >> 5;
  const int lane = tid & 31;
  const int hg   = lane >> 4;
  const int nn   = lane & 15;

  const int qb   = blockIdx.x * 4 + wave;
  const int i0   = (qb & 127) * 16;
  const int bh   = qb >> 7;
  const int bI   = bh >> 4;
  const int hIdx = bh & 15;

  int halfw = wsz[0] >> 1;
  if (halfw > 128) halfw = 128;
  const int js = i0 - ((halfw + 15) & ~15);
  int T = (((i0 + 15 + halfw) - js) >> 4) + 1;
  if (T & 1) T++;
  if (T > MAXT) T = MAXT;

  const size_t base = (size_t)bh * N_ * HD_;

  // Q A-fragments (d = [0,32) and [32,64)); 16B-aligned vector loads
  v16h qa[2];
  {
    const _Float16* p = qh + base + (size_t)(i0 + nn) * HD_;
    #pragma unroll
    for (int g = 0; g < 2; ++g)
      qa[g] = cat8(*(const v8h*)(p + 32 * g + hg * 8),
                   *(const v8h*)(p + 32 * g + 16 + hg * 8));
  }

  float rmax[8], rsum[8];
  #pragma unroll
  for (int r = 0; r < 8; ++r) { rmax[r] = -3.0e38f; rsum[r] = 0.0f; }

  // ---- pass 1: scores + online softmax statistics
  for (int t = 0; t < T; ++t) {
    const int j0 = js + t * 16;
    const int jj = j0 + nn;
    v8f s = {};
    {
      const int jc = jj < 0 ? 0 : (jj > N_ - 1 ? N_ - 1 : jj);
      const _Float16* p = kh + base + (size_t)jc * HD_;
      #pragma unroll
      for (int g = 0; g < 2; ++g) {
        const v16h kb = cat8(*(const v8h*)(p + 32 * g + hg * 8),
                             *(const v8h*)(p + 32 * g + 16 + hg * 8));
        s = wmma_f32_f16(qa[g], kb, s);
      }
    }
    const bool jin   = (jj >= 0) && (jj < N_);
    const bool jmask = jin && (mask[(size_t)bI * N_ + (jin ? jj : 0)] != 0);
    #pragma unroll
    for (int r = 0; r < 8; ++r) {
      const int iQ = i0 + r + 8 * hg;
      float sv = s[r] * 0.125f;
      const bool ok = jmask && (jj >= iQ - halfw) && (jj <= iQ + halfw);
      const _Float16 sh = ok ? (_Float16)sv : (_Float16)NEG_LIM;
      sS[wave][r + 8 * hg][t * 16 + nn] = sh;
      const float svr = (float)sh;
      float mloc = svr;
      #pragma unroll
      for (int o = 1; o < 16; o <<= 1) mloc = fmaxf(mloc, __shfl_xor(mloc, o, 32));
      const float nm = fmaxf(rmax[r], mloc);
      float ps = __expf(svr - nm);
      #pragma unroll
      for (int o = 1; o < 16; o <<= 1) ps += __shfl_xor(ps, o, 32);
      rsum[r] = rsum[r] * __expf(rmax[r] - nm) + ps;
      rmax[r] = nm;
    }
  }
  #pragma unroll
  for (int r = 0; r < 8; ++r) {
    sMax[wave][r + 8 * hg] = rmax[r];
    sSum[wave][r + 8 * hg] = rsum[r];
  }
  __syncthreads();

  // ---- pass 2: O = P @ V
  v8f o4[4] = {};
  const float rmQ = sMax[wave][nn];
  for (int t2 = 0; t2 < T; t2 += 2) {
    v16h pa;
    #pragma unroll
    for (int g = 0; g < 2; ++g) {
      const v8h sraw = *(const v8h*)&sS[wave][nn][t2 * 16 + hg * 16 + g * 8];
      #pragma unroll
      for (int w = 0; w < 8; ++w)
        pa[g * 8 + w] = (_Float16)__expf((float)sraw[w] - rmQ);
    }
    const int jv0 = js + t2 * 16 + hg * 16;
    const bool infast = (jv0 >= 0) && (jv0 + 15 < N_);
    #pragma unroll
    for (int a4 = 0; a4 < 4; ++a4) {
      const int dIdx = a4 * 16 + nn;
      const _Float16* pv = vt + base + (size_t)dIdx * N_;
      v16h vb;
      if (infast) {
        vb = cat8(*(const v8h*)(pv + jv0), *(const v8h*)(pv + jv0 + 8));
      } else {
        #pragma unroll
        for (int tt = 0; tt < 16; ++tt) {
          const int jv = jv0 + tt;
          const int jc = jv < 0 ? 0 : (jv > N_ - 1 ? N_ - 1 : jv);
          vb[tt] = pv[jc];                 // p==0 for out-of-window keys
        }
      }
      o4[a4] = wmma_f32_f16(pa, vb, o4[a4]);
    }
  }

  #pragma unroll
  for (int a4 = 0; a4 < 4; ++a4) {
    const int dIdx = a4 * 16 + nn;
    #pragma unroll
    for (int r = 0; r < 8; ++r) {
      const int m   = r + 8 * hg;
      const int tok = i0 + m;
      const float val = o4[a4][r] / sSum[wave][m];
      ao[((size_t)bI * N_ + tok) * INNER_ + hIdx * HD_ + dIdx] = (_Float16)val;
    }
  }
}

// ---------------------------------------------------------------------------
// Kernel 3: output projection. A operand is f16 already -> stage it with the
// CDNA5 async DMA path (global_load_async_to_lds_b128 + s_wait_asynccnt).
// ---------------------------------------------------------------------------
__global__ __launch_bounds__(256)
void oproj_kernel(const _Float16* __restrict__ Aw, const float* __restrict__ Wo,
                  const float* __restrict__ bo,
                  const unsigned char* __restrict__ mask,
                  float* __restrict__ out)
{
  __shared__ __align__(16) _Float16 As[64 * LDA];
  __shared__ __align__(16) _Float16 Bs[128 * LDB];

  const int row0 = blockIdx.x * 64;
  const int col0 = blockIdx.y * 128;
  const int tid  = threadIdx.x;
  const int wave = tid >> 5, lane = tid & 31;
  const int hg   = lane >> 4, nn = lane & 15;
  const int wm   = wave >> 1, wn = wave & 1;

  const int bk_ = (tid >> 5) * 4;
  const int bn_ = (tid & 31) * 4;

  // per-thread A staging addresses (A is f16: 16B per thread, no conversion)
  const int ae = tid * 8;
  const int ar = ae >> 5, ac = ae & 31;
  const unsigned a_loff = (unsigned)(size_t)&As[ar * LDA + ac];

  v8f acc[4] = {};
  for (int k0 = 0; k0 < INNER_; k0 += 32) {
    __syncthreads();
    // ---- async DMA: 128b per lane, memory -> LDS, tracked by ASYNCcnt
    {
      const unsigned long long ga =
          (unsigned long long)(size_t)(Aw + (size_t)(row0 + ar) * INNER_ + k0 + ac);
      asm volatile("global_load_async_to_lds_b128 %0, %1, off"
                   :: "v"(a_loff), "v"(ga) : "memory");
    }
    // ---- stage B: Wo tile 32x128 f32 -> f16 transposed [n][k]
    {
      float rr[4][4];
      #pragma unroll
      for (int i = 0; i < 4; ++i) {
        float4 v = *(const float4*)(Wo + (size_t)(k0 + bk_ + i) * D_ + col0 + bn_);
        rr[i][0] = v.x; rr[i][1] = v.y; rr[i][2] = v.z; rr[i][3] = v.w;
      }
      #pragma unroll
      for (int j = 0; j < 4; ++j) {
        v4h pk = {(_Float16)rr[0][j], (_Float16)rr[1][j],
                  (_Float16)rr[2][j], (_Float16)rr[3][j]};
        *(v4h*)&Bs[(bn_ + j) * LDB + bk_] = pk;
      }
    }
    if (k0 + 32 < INNER_)
      __builtin_prefetch(Wo + (size_t)(k0 + 32) * D_ + col0, 0, 3);
    asm volatile("s_wait_asynccnt 0x0" ::: "memory");
    __syncthreads();

    const _Float16* pa = &As[(wm * 16 + nn) * LDA + hg * 8];
    const v16h afr = cat8(*(const v8h*)pa, *(const v8h*)(pa + 16));
    #pragma unroll
    for (int a4 = 0; a4 < 4; ++a4) {
      const _Float16* pb = &Bs[(wn * 64 + a4 * 16 + nn) * LDB + hg * 8];
      const v16h bfr = cat8(*(const v8h*)pb, *(const v8h*)(pb + 16));
      acc[a4] = wmma_f32_f16(afr, bfr, acc[a4]);
    }
  }

  #pragma unroll
  for (int a4 = 0; a4 < 4; ++a4) {
    const int C  = col0 + wn * 64 + a4 * 16 + nn;
    const float bc = bo[C];
    #pragma unroll
    for (int r = 0; r < 8; ++r) {
      const int R   = row0 + r + 8 * hg;
      const int bI  = R >> 11;
      const int tok = R & (N_ - 1);
      float val = acc[a4][r] + bc;
      if (!mask[(size_t)bI * N_ + tok]) val = 0.0f;
      out[(size_t)R * D_ + C] = val;
    }
  }
}

// ---------------------------------------------------------------------------
extern "C" void kernel_launch(void* const* d_in, const int* in_sizes, int n_in,
                              void* d_out, int out_size, void* d_ws, size_t ws_size,
                              hipStream_t stream)
{
  (void)in_sizes; (void)n_in; (void)out_size; (void)ws_size;
  const float* x  = (const float*)d_in[0];
  const unsigned char* mask = (const unsigned char*)d_in[1];
  const float* freqs = (const float*)d_in[2];
  const float* Wq = (const float*)d_in[3];  const float* bq = (const float*)d_in[4];
  const float* Wk = (const float*)d_in[5];  const float* bk = (const float*)d_in[6];
  const float* Wv = (const float*)d_in[7];  const float* bv = (const float*)d_in[8];
  const float* Wo = (const float*)d_in[9];  const float* bo = (const float*)d_in[10];
  const int*  wsz = (const int*)d_in[11];
  float* out = (float*)d_out;

  const size_t per = (size_t)B_ * H_ * N_ * HD_;   // 4,194,304 f16 elems
  _Float16* qh = (_Float16*)d_ws;
  _Float16* kh = qh + per;
  _Float16* vt = kh + per;
  _Float16* ao = vt + per;

  qkv_rope_kernel<<<dim3(64, 8, 3), 256, 0, stream>>>(
      x, Wq, bq, Wk, bk, Wv, bv, freqs, qh, kh, vt);
  attn_kernel<<<dim3((B_ * H_ * (N_ / 16)) / 4), 128, 0, stream>>>(
      qh, kh, vt, mask, wsz, ao);
  oproj_kernel<<<dim3(64, 8), 256, 0, stream>>>(ao, Wo, bo, mask, out);
}
